// NeRFMLPNetwork_24275155157282
// MI455X (gfx1250) — matrix operand order repaired and checked by reading, not compiled, not run on hardware
//
#include <hip/hip_runtime.h>
#include <hip/hip_bf16.h>

typedef __bf16 bf16x8 __attribute__((ext_vector_type(8)));
typedef __bf16 v16bf  __attribute__((ext_vector_type(16)));
typedef float  v8f    __attribute__((ext_vector_type(8)));

#define WDIM 512
#define HWSZ 32768           // 64*512
#define PITCH 136            // bf16 elements per activation row (pad vs 128)
#define FPITCH 68            // f32 elements per output-staging row (64ch half)
#define GAIN_P 1.4142135623730951f
#define GAIN_N 0.28284271247461901f   // 0.2 * sqrt(2)

// ---------------------------------------------------------------------------
// Kernel 1: fold style modulation + demod into per-(block,batch) bf16 weights
//   wmod layout: block0: [b][128][64]  at offset 0           (4*8192 elems)
//                blk k : [k-1][b][128][128] at offset 32768
// ---------------------------------------------------------------------------
__global__ __launch_bounds__(256) void modw_kernel(
    const float* __restrict__ wp,        // [4, 8, 512]
    const float* __restrict__ aff_w_in,  // [60, 512]
    const float* __restrict__ aff_b_in,  // [60]
    const float* __restrict__ conv_w_in, // [128, 60]
    const float* __restrict__ aff_w,     // [7, 128, 512]
    const float* __restrict__ aff_b,     // [7, 128]
    const float* __restrict__ conv_w,    // [7, 128, 128]
    __bf16* __restrict__ wmod)
{
    const int blk = blockIdx.x >> 2;
    const int b   = blockIdx.x & 3;
    const int cin  = (blk == 0) ? 60 : 128;
    const int kpad = (blk == 0) ? 64 : 128;
    const float* awp = (blk == 0) ? aff_w_in  : aff_w  + (size_t)(blk-1)*128*WDIM;
    const float* abp = (blk == 0) ? aff_b_in  : aff_b  + (blk-1)*128;
    const float* cwp = (blk == 0) ? conv_w_in : conv_w + (size_t)(blk-1)*128*128;
    const float* wpb = wp + (size_t)(b*8 + blk)*WDIM;
    __bf16* wout = (blk == 0) ? wmod + (size_t)b*128*64
                              : wmod + 32768 + (size_t)((blk-1)*4 + b)*128*128;

    __shared__ float s_sh[128];
    __shared__ float d_sh[128];
    const int t = threadIdx.x;

    if (t < cin) {                       // style: s = wp . aw[c] / sqrt(512) + ab[c]
        const float4* wv = (const float4*)wpb;
        const float4* av = (const float4*)(awp + (size_t)t*WDIM);
        float acc = 0.f;
        #pragma unroll 4
        for (int i = 0; i < WDIM/4; ++i) {
            float4 a = av[i], w = wv[i];
            acc += a.x*w.x + a.y*w.y + a.z*w.z + a.w*w.w;
        }
        s_sh[t] = acc * 0.044194173824159216f + abp[t];
    }
    __syncthreads();
    if (t < 128) {                       // demod: d = rsqrt(sum (cw*s)^2 + 1e-8)
        float acc = 1e-8f;
        for (int i = 0; i < cin; ++i) {
            float v = cwp[t*cin + i] * s_sh[i];
            acc += v*v;
        }
        d_sh[t] = rsqrtf(acc);
    }
    __syncthreads();
    const int total = 128 * kpad;        // folded weight: W[o][i] = cw*s[i]*d[o]
    for (int idx = t; idx < total; idx += 256) {
        int o = idx >> ((blk == 0) ? 6 : 7);
        int i = idx & (kpad - 1);
        float v = (i < cin) ? cwp[o*cin + i] * s_sh[i] * d_sh[o] : 0.f;
        wout[idx] = (__bf16)v;
    }
}

// ---------------------------------------------------------------------------
// Kernel 2: fused 8-block modulated-conv chain via v_wmma_f32_16x16x32_bf16
//   WG = 256 thr (8 waves), 128 pixels; wave w owns pixels [16w,16w+16)
// ---------------------------------------------------------------------------
__global__ __launch_bounds__(256) void synth_kernel(
    const float*  __restrict__ x,     // [4, 60, 32768]
    const float*  __restrict__ cb0,   // [128]
    const float*  __restrict__ cbk,   // [7, 128]
    const __bf16* __restrict__ wmod,
    float*        __restrict__ out)   // [4, 128, 32768]
{
    __shared__ __align__(16) __bf16 act[8 * 16 * PITCH];   // 34816 B
    const int tid = threadIdx.x;
    const int b   = blockIdx.y;
    const int p0  = blockIdx.x * 128;

    // ---- load x tile -> LDS [pixel][channel] bf16, 60 real + 4 zero channels
    for (int idx = tid; idx < 64*128; idx += 256) {
        int c = idx >> 7;                 // 0..63
        int p = idx & 127;                // coalesced along pixels
        float v = (c < 60) ? x[((size_t)(b*60 + c) << 15) + p0 + p] : 0.f;
        act[(p >> 4)*(16*PITCH) + (p & 15)*PITCH + c] = (__bf16)v;
    }
    __syncthreads();

    const int lane = tid & 31;
    const int wave = tid >> 5;
    const int hl   = lane >> 4;           // half-lane group
    const int m    = lane & 15;           // A-row / also C/D column id
    __bf16* myact  = act + wave*(16*PITCH);

    v8f acc[8];

    // ================= block 0 (K = 64) =================
    {
        const __bf16* W = wmod + (size_t)b*(128*64);
        v16bf a[2];
        #pragma unroll
        for (int kt = 0; kt < 2; ++kt) {  // 16-bit A layout: k = kt*32 + hl*8 + {0..7,16..23}
            const __bf16* base = myact + m*PITCH + kt*32 + hl*8;
            ((bf16x8*)&a[kt])[0] = *(const bf16x8*)(base);
            ((bf16x8*)&a[kt])[1] = *(const bf16x8*)(base + 16);
        }
        #pragma unroll
        for (int nt = 0; nt < 8; ++nt) {
            v8f c = {};
            #pragma unroll
            for (int kt = 0; kt < 2; ++kt) {
                v16bf bm;                 // B layout: n = lane&15, k = kt*32 + hl*8 + {0..7,16..23}
                const __bf16* wb = W + (nt*16 + m)*64 + kt*32 + hl*8;
                ((bf16x8*)&bm)[0] = *(const bf16x8*)(wb);
                ((bf16x8*)&bm)[1] = *(const bf16x8*)(wb + 16);
                c = __builtin_amdgcn_wmma_f32_16x16x32_bf16(
                        false, a[kt], false, bm, (short)0, c, false, false);
            }
            acc[nt] = c;
        }
        #pragma unroll
        for (int nt = 0; nt < 8; ++nt) {  // bias + lrelu*sqrt2 -> bf16 act
            float bias = cb0[nt*16 + m];
            #pragma unroll
            for (int v = 0; v < 8; ++v) { // C/D layout: row = v + 8*hl, col = lane&15
                float y = acc[nt][v] + bias;
                y *= (y >= 0.f) ? GAIN_P : GAIN_N;
                myact[(v + hl*8)*PITCH + nt*16 + m] = (__bf16)y;
            }
        }
    }

    // ================= blocks 1..7 (K = 128) =================
    for (int blk = 1; blk < 8; ++blk) {
        asm volatile("" ::: "memory");    // keep DS store->load program order
        const __bf16* W  = wmod + 32768 + (size_t)((blk-1)*4 + b)*(128*128);
        const float*  cb = cbk + (blk-1)*128;
        if (blk < 7) {                    // prefetch next block's weights into L2
            const __bf16* Wn = wmod + 32768 + (size_t)(blk*4 + b)*(128*128);
            __builtin_prefetch(Wn + (size_t)tid*64, 0, 1);
        }
        v16bf a[4];
        #pragma unroll
        for (int kt = 0; kt < 4; ++kt) {
            const __bf16* base = myact + m*PITCH + kt*32 + hl*8;
            ((bf16x8*)&a[kt])[0] = *(const bf16x8*)(base);
            ((bf16x8*)&a[kt])[1] = *(const bf16x8*)(base + 16);
        }
        #pragma unroll
        for (int nt = 0; nt < 8; ++nt) {
            v8f c = {};
            #pragma unroll
            for (int kt = 0; kt < 4; ++kt) {
                v16bf bm;
                const __bf16* wb = W + (nt*16 + m)*128 + kt*32 + hl*8;
                ((bf16x8*)&bm)[0] = *(const bf16x8*)(wb);
                ((bf16x8*)&bm)[1] = *(const bf16x8*)(wb + 16);
                c = __builtin_amdgcn_wmma_f32_16x16x32_bf16(
                        false, a[kt], false, bm, (short)0, c, false, false);
            }
            acc[nt] = c;
        }
        if (blk < 7) {
            #pragma unroll
            for (int nt = 0; nt < 8; ++nt) {
                float bias = cb[nt*16 + m];
                #pragma unroll
                for (int v = 0; v < 8; ++v) {
                    float y = acc[nt][v] + bias;
                    y *= (y >= 0.f) ? GAIN_P : GAIN_N;
                    myact[(v + hl*8)*PITCH + nt*16 + m] = (__bf16)y;
                }
            }
        }
    }

    // ================= output: stage f32 through LDS, coalesced store =======
    const float* cb7 = cbk + 6*128;
    float* fout  = (float*)act;                 // reuse act buffer (same size)
    float* myout = fout + wave*(16*FPITCH);
    #pragma unroll
    for (int half = 0; half < 2; ++half) {      // 64 channels per pass
        #pragma unroll
        for (int nt4 = 0; nt4 < 4; ++nt4) {
            int nt = half*4 + nt4;
            float bias = cb7[nt*16 + m];
            #pragma unroll
            for (int v = 0; v < 8; ++v) {
                float y = acc[nt][v] + bias;
                y *= (y >= 0.f) ? GAIN_P : GAIN_N;
                myout[(v + hl*8)*FPITCH + nt4*16 + m] = y;
            }
        }
        __syncthreads();
        for (int idx = tid; idx < 64*128; idx += 256) {
            int c = idx >> 7;                   // 0..63
            int p = idx & 127;                  // coalesced along pixels
            out[((size_t)(b*128 + half*64 + c) << 15) + p0 + p] =
                fout[(p >> 4)*(16*FPITCH) + (p & 15)*FPITCH + c];
        }
        __syncthreads();
    }
}

// ---------------------------------------------------------------------------
extern "C" void kernel_launch(void* const* d_in, const int* in_sizes, int n_in,
                              void* d_out, int out_size, void* d_ws, size_t ws_size,
                              hipStream_t stream) {
    const float* x          = (const float*)d_in[1];   // points_encoding
    const float* wp         = (const float*)d_in[2];
    const float* aff_w_in   = (const float*)d_in[3];
    const float* aff_b_in   = (const float*)d_in[4];
    const float* conv_w_in  = (const float*)d_in[5];
    const float* conv_b_in  = (const float*)d_in[6];
    const float* aff_w      = (const float*)d_in[7];
    const float* aff_b      = (const float*)d_in[8];
    const float* conv_w     = (const float*)d_in[9];
    const float* conv_b     = (const float*)d_in[10];
    __bf16* wmod = (__bf16*)d_ws;   // 32768 + 7*4*16384 bf16 = 960 KB

    modw_kernel<<<32, 256, 0, stream>>>(wp, aff_w_in, aff_b_in, conv_w_in,
                                        aff_w, aff_b, conv_w, wmod);
    dim3 grid(HWSZ / 128, 4);
    synth_kernel<<<grid, 256, 0, stream>>>(x, conv_b_in, conv_b, wmod,
                                           (float*)d_out);
}
